// _GRUFallback_71923522339388
// MI455X (gfx1250) — matrix operand order repaired and checked by reading, not compiled, not run on hardware
//
#include <hip/hip_runtime.h>
#include <hip/hip_bf16.h>
#include <hip/hip_fp16.h>

#define B_   32
#define T_   512
#define D_   1024
#define G_   3072            // 3*D
#define BT_  (B_ * T_)       // 16384
#define REC_BLOCKS 48        // 48 blocks * 8 waves = 384 tiles = (32/16)*(3072/16)

typedef __attribute__((ext_vector_type(16))) __bf16        v16bf;
typedef __attribute__((ext_vector_type(8)))  float         v8f;
typedef __attribute__((ext_vector_type(8)))  unsigned short ush8;

// ---------------------------------------------------------------------------
// WMMA fragment loaders (bf16 16x16x32, wave32 layouts per CDNA5 ISA 7.12.2)
// A: 16x32 (MxK). lane<16 -> M=lane, K={k0..k0+7, k0+16..k0+23}
//                 lane>=16 -> M=lane-16, K={k0+8..k0+15, k0+24..k0+31}
// ---------------------------------------------------------------------------
__device__ __forceinline__ v16bf load_frag_a(const __hip_bfloat16* base, int ld,
                                             int m0, int k0, int lane) {
    int row = m0 + (lane & 15);
    int kk  = k0 + ((lane >> 4) << 3);                 // +8 for upper half-wave
    const __hip_bfloat16* p = base + (size_t)row * ld + kk;
    union { ush8 h[2]; v16bf v; } u;
    u.h[0] = *(const ush8*)(p);                        // K = kk .. kk+7
    u.h[1] = *(const ush8*)(p + 16);                   // K = kk+16 .. kk+23
    return u.v;
}

// B: 32x16 (KxN), weights stored row-major (N,K) so B[k][n] = W[n*ld + k].
// lane<16 -> N=n0+lane,   K={k0..k0+7, k0+16..k0+23} ; lane>=16 -> N=n0+lane-16, K={k0+8..k0+15, k0+24..k0+31}
__device__ __forceinline__ v16bf load_frag_b(const __hip_bfloat16* w, int ld,
                                             int n0, int k0, int lane) {
    int col = n0 + (lane & 15);
    int kk  = k0 + ((lane >> 4) << 3);                 // +8 for upper half-wave
    const __hip_bfloat16* p = w + (size_t)col * ld + kk;
    union { ush8 h[2]; v16bf v; } u;
    u.h[0] = *(const ush8*)(p);
    u.h[1] = *(const ush8*)(p + 16);
    return u.v;
}

// ---------------------------------------------------------------------------
// Grid-wide sense-reversing barrier (device-scope atomics in global memory)
// ---------------------------------------------------------------------------
__device__ __forceinline__ void grid_barrier(unsigned* bar, unsigned nblocks) {
    __syncthreads();
    if (threadIdx.x == 0) {
        __threadfence();
        unsigned gen  = __hip_atomic_load(&bar[1], __ATOMIC_RELAXED, __HIP_MEMORY_SCOPE_AGENT);
        unsigned prev = __hip_atomic_fetch_add(&bar[0], 1u, __ATOMIC_ACQ_REL, __HIP_MEMORY_SCOPE_AGENT);
        if (prev == nblocks - 1u) {
            __hip_atomic_store(&bar[0], 0u, __ATOMIC_RELAXED, __HIP_MEMORY_SCOPE_AGENT);
            __hip_atomic_fetch_add(&bar[1], 1u, __ATOMIC_RELEASE, __HIP_MEMORY_SCOPE_AGENT);
        } else {
            while (__hip_atomic_load(&bar[1], __ATOMIC_ACQUIRE, __HIP_MEMORY_SCOPE_AGENT) == gen)
                __builtin_amdgcn_s_sleep(1);
        }
    }
    __syncthreads();
}

// ---------------------------------------------------------------------------
// f32 -> bf16 conversion (grid-stride)
// ---------------------------------------------------------------------------
__global__ void cvt_f32_bf16(const float* __restrict__ src,
                             __hip_bfloat16* __restrict__ dst, int n) {
    int i = blockIdx.x * blockDim.x + threadIdx.x;
    int stride = gridDim.x * blockDim.x;
    for (; i < n; i += stride) dst[i] = __float2bfloat16(src[i]);
}

// Init hidden state (f32 + bf16 mirror) and zero barrier counters
__global__ void init_state(const float* __restrict__ h0, float* hf,
                           __hip_bfloat16* hb, unsigned* bar, int n) {
    int gtid = blockIdx.x * blockDim.x + threadIdx.x;
    if (gtid < 2) bar[gtid] = 0u;
    int stride = gridDim.x * blockDim.x;
    for (int i = gtid; i < n; i += stride) {
        float v = h0[i];
        hf[i] = v;
        hb[i] = __float2bfloat16(v);
    }
}

// ---------------------------------------------------------------------------
// Phase 1: x_proj[bt, g] = x[bt, :] . w_ih[g, :] + b_ih[g], stored bf16.
// 2x2 register-blocked: one 32x32 macro-tile (4 WMMA accumulators) per wave.
// Halves A and B L2 traffic vs 1-tile/wave; exact tiling -> EXEC all 1s.
// ---------------------------------------------------------------------------
__global__ void __launch_bounds__(256)
xproj_wmma(const __hip_bfloat16* __restrict__ xb,
           const __hip_bfloat16* __restrict__ wih,
           const float* __restrict__ bih,
           __hip_bfloat16* __restrict__ xproj) {
    const int lane  = threadIdx.x & 31;
    const int wave  = threadIdx.x >> 5;
    const int macro = blockIdx.x * 8 + wave;         // 0 .. 512*96-1
    const int ntm = macro % (G_ / 32);
    const int mtm = macro / (G_ / 32);
    const int m0 = mtm * 32, n0 = ntm * 32;

    v8f c00 = {}, c01 = {}, c10 = {}, c11 = {};
    #pragma unroll 2
    for (int k0 = 0; k0 < D_; k0 += 32) {
        v16bf a0 = load_frag_a(xb,  D_, m0,      k0, lane);
        v16bf a1 = load_frag_a(xb,  D_, m0 + 16, k0, lane);
        v16bf b0 = load_frag_b(wih, D_, n0,      k0, lane);
        v16bf b1 = load_frag_b(wih, D_, n0 + 16, k0, lane);
        c00 = __builtin_amdgcn_wmma_f32_16x16x32_bf16(false, a0, false, b0, (short)0, c00, false, false);
        c01 = __builtin_amdgcn_wmma_f32_16x16x32_bf16(false, a0, false, b1, (short)0, c01, false, false);
        c10 = __builtin_amdgcn_wmma_f32_16x16x32_bf16(false, a1, false, b0, (short)0, c10, false, false);
        c11 = __builtin_amdgcn_wmma_f32_16x16x32_bf16(false, a1, false, b1, (short)0, c11, false, false);
    }
    const int col = lane & 15;
    const int rb  = (lane >> 4) << 3;                // D-layout: lanes>=16 hold M=8..15
    const float bias0 = bih[n0 + col];
    const float bias1 = bih[n0 + 16 + col];
    #pragma unroll
    for (int r = 0; r < 8; ++r) {
        size_t ro0 = (size_t)(m0 + rb + r) * G_;
        size_t ro1 = (size_t)(m0 + 16 + rb + r) * G_;
        xproj[ro0 + n0 + col]      = __float2bfloat16(c00[r] + bias0);
        xproj[ro0 + n0 + 16 + col] = __float2bfloat16(c01[r] + bias1);
        xproj[ro1 + n0 + col]      = __float2bfloat16(c10[r] + bias0);
        xproj[ro1 + n0 + 16 + col] = __float2bfloat16(c11[r] + bias1);
    }
}

// ---------------------------------------------------------------------------
// Phase 2: persistent recurrent kernel. 48 blocks, 8 waves each = 384 waves,
// one 16x16 tile of hp = h . w_hh^T + b_hh per wave per step.
// w_hh B-fragments for this wave's tile column are preloaded into 256 VGPRs
// once (occupancy is ~1 wave/SIMD, so the VGPR budget is free) so the serial
// per-step path is only 32 A-frag loads (h, L2-resident) + 32 WMMAs.
// ---------------------------------------------------------------------------
__global__ void __launch_bounds__(256)
gru_recurrent(const __hip_bfloat16* __restrict__ xproj,
              const __hip_bfloat16* __restrict__ whh,
              const float* __restrict__ bhh,
              const int* __restrict__ dones,
              float* hp_buf, float* hf, __hip_bfloat16* hb,
              float* out, unsigned* bar) {
    const int lane = threadIdx.x & 31;
    const int wave = threadIdx.x >> 5;
    const int tile = blockIdx.x * 8 + wave;          // 0..383
    const int mt = tile & 1;                         // 2 M-tiles (B=32)
    const int nt = tile >> 1;                        // 192 N-tiles (G=3072)
    const int m0 = mt * 16, n0 = nt * 16;
    const int col = lane & 15;
    const int rb  = (lane >> 4) << 3;
    const float biasc = bhh[n0 + col];
    const int gtid = blockIdx.x * blockDim.x + threadIdx.x;
    const int nthr = gridDim.x * blockDim.x;
    const unsigned nblocks = gridDim.x;

    // Step-invariant w_hh fragments: keep resident in VGPRs for all 512 steps.
    v16bf wbfrag[32];
    #pragma unroll
    for (int kk = 0; kk < 32; ++kk)
        wbfrag[kk] = load_frag_b(whh, D_, n0, kk * 32, lane);

    for (int t = 0; t < T_; ++t) {
        // ---- hp tile: 32 WMMAs over K=1024, accumulator seeded with b_hh ----
        v8f c;
        #pragma unroll
        for (int r = 0; r < 8; ++r) c[r] = biasc;
        #pragma unroll
        for (int kk = 0; kk < 32; ++kk) {
            v16bf a = load_frag_a(hb, D_, m0, kk * 32, lane);
            c = __builtin_amdgcn_wmma_f32_16x16x32_bf16(false, a, false, wbfrag[kk],
                                                        (short)0, c, false, false);
        }
        #pragma unroll
        for (int r = 0; r < 8; ++r)
            hp_buf[(m0 + rb + r) * G_ + n0 + col] = c[r];

        grid_barrier(bar, nblocks);

        // ---- fused gates + done-masked hidden update ----
        for (int i = gtid; i < B_ * D_; i += nthr) {
            int b = i >> 10;                         // / D_
            int d = i & (D_ - 1);
            size_t xo = (size_t)(b * T_ + t) * G_;
            float xr = __bfloat162float(xproj[xo + d]);
            float xz = __bfloat162float(xproj[xo + D_ + d]);
            float xn = __bfloat162float(xproj[xo + 2 * D_ + d]);
            int ho = b * G_;
            float hr = hp_buf[ho + d];
            float hz = hp_buf[ho + D_ + d];
            float hn = hp_buf[ho + 2 * D_ + d];
            float r  = 1.0f / (1.0f + __expf(-(xr + hr)));
            float z  = 1.0f / (1.0f + __expf(-(xz + hz)));
            float nn = tanhf(xn + r * hn);
            float hprev = hf[i];
            float hnew  = (1.0f - z) * nn + z * hprev;
            out[(size_t)(b * T_ + t) * D_ + d] = hnew;
            float carry = hnew;
            if (t < T_ - 1 && dones[b * T_ + t] != 0) carry = 0.0f;
            hf[i] = carry;
            hb[i] = __float2bfloat16(carry);
            if (t == T_ - 1) out[(size_t)BT_ * D_ + i] = carry;  // h_final tail
            if (t + 1 < T_) __builtin_prefetch(&xproj[xo + G_ + d], 0, 0);
        }

        grid_barrier(bar, nblocks);
    }
}

// ---------------------------------------------------------------------------
extern "C" void kernel_launch(void* const* d_in, const int* in_sizes, int n_in,
                              void* d_out, int out_size, void* d_ws, size_t ws_size,
                              hipStream_t stream) {
    const float* x     = (const float*)d_in[0];   // (B,T,D)
    const float* h0    = (const float*)d_in[1];   // (1,B,D)
    const int*   dones = (const int*)  d_in[2];   // (B,T)
    const float* wih   = (const float*)d_in[3];   // (3D,D)
    const float* whh   = (const float*)d_in[4];   // (3D,D)
    const float* bih   = (const float*)d_in[5];   // (3D,)
    const float* bhh   = (const float*)d_in[6];   // (3D,)
    float* out = (float*)d_out;                   // (B,T,D) ++ (1,B,D)

    char* ws = (char*)d_ws;
    size_t off = 0;
    auto alloc = [&](size_t bytes) -> void* {
        void* p = ws + off;
        off += (bytes + 255) & ~(size_t)255;
        return p;
    };
    unsigned*        bar    = (unsigned*)       alloc(256);
    __hip_bfloat16*  xb     = (__hip_bfloat16*) alloc((size_t)BT_ * D_ * 2);   // 32 MB
    __hip_bfloat16*  wihb   = (__hip_bfloat16*) alloc((size_t)G_  * D_ * 2);   // 6 MB
    __hip_bfloat16*  whhb   = (__hip_bfloat16*) alloc((size_t)G_  * D_ * 2);   // 6 MB
    __hip_bfloat16*  xproj  = (__hip_bfloat16*) alloc((size_t)BT_ * G_ * 2);   // 96 MB
    float*           hp_buf = (float*)          alloc((size_t)B_  * G_ * 4);
    float*           hf     = (float*)          alloc((size_t)B_  * D_ * 4);
    __hip_bfloat16*  hb     = (__hip_bfloat16*) alloc((size_t)B_  * D_ * 2);
    (void)in_sizes; (void)n_in; (void)out_size; (void)ws_size;

    // Phase 0: precision conversion + state/barrier init
    cvt_f32_bf16<<<1024, 256, 0, stream>>>(x,   xb,   BT_ * D_);
    cvt_f32_bf16<<<256,  256, 0, stream>>>(wih, wihb, G_ * D_);
    cvt_f32_bf16<<<256,  256, 0, stream>>>(whh, whhb, G_ * D_);
    init_state  <<<64,   256, 0, stream>>>(h0, hf, hb, bar, B_ * D_);

    // Phase 1: x-projection GEMM (16384x3072x1024), 32x32 macro-tile per wave
    xproj_wmma<<<(BT_ / 32) * (G_ / 32) / 8, 256, 0, stream>>>(xb, wihb, bih, xproj);

    // Phase 2: persistent recurrence across T=512 steps
    gru_recurrent<<<REC_BLOCKS, 256, 0, stream>>>(xproj, whhb, bhh, dones,
                                                  hp_buf, hf, hb, out, bar);
}
